// SimVQ_2654289789559
// MI455X (gfx1250) — matrix operand order repaired and checked by reading, not compiled, not run on hardware
//
#include <hip/hip_runtime.h>
#include <hip/hip_bf16.h>

#define DIMD  512
#define NCODE 8192
#define NTOK  8192   // B*N = 4*2048
#define XROWS 520    // padded LDS row stride in bf16 elems (1040 B = 65*16)

typedef __attribute__((ext_vector_type(16))) __bf16       bf16x16;
typedef __attribute__((ext_vector_type(8)))  float        f32x8;
typedef __attribute__((ext_vector_type(4)))  unsigned int uint4v;

union FragU { bf16x16 v; uint4v q[2]; };

__device__ __forceinline__ f32x8 wmma_bf16(const FragU& a, const FragU& b, f32x8 c) {
  // D = A(16x32 bf16) * B(32x16 bf16) + C(16x16 f32)
  return __builtin_amdgcn_wmma_f32_16x16x32_bf16(false, a.v, false, b.v,
                                                 (short)0, c, false, false);
}

// ---------------------------------------------------------------- converts
__global__ void split_f32_to_bf16(const float* __restrict__ in,
                                  unsigned short* __restrict__ hi,
                                  unsigned short* __restrict__ lo, int n) {
  int i = blockIdx.x * blockDim.x + threadIdx.x;
  int stride = gridDim.x * blockDim.x;
  for (; i < n; i += stride) {
    float x = in[i];
    unsigned int u = __float_as_uint(x) & 0xFFFF0000u;   // truncated hi bf16
    hi[i] = (unsigned short)(u >> 16);
    if (lo) {
      float r = x - __uint_as_float(u);                  // residual
      lo[i] = (unsigned short)(__float_as_uint(r) >> 16);
    }
  }
}

// ---------------------------------------------------------------- GEMM 1
// implicit[c,e] = sum_d codebook[c,d] * W[e,d]   (NT, both row-major along K)
// split-bf16: AB ~= Ah*Bh + Ah*Bl + Al*Bh  (near-fp32 accuracy)
__global__ __launch_bounds__(128) void gemm1_implicit(
    const unsigned short* __restrict__ cb_hi, const unsigned short* __restrict__ cb_lo,
    const unsigned short* __restrict__ w_hi,  const unsigned short* __restrict__ w_lo,
    float* __restrict__ impl, unsigned short* __restrict__ imp_hi) {
  const int wave = threadIdx.x >> 5;
  const int lane = threadIdx.x & 31;
  const int g  = lane >> 4;       // half-wave group
  const int lr = lane & 15;
  const int t = blockIdx.x * 4 + wave;     // 0..4095 wave tiles
  const int tileM = (t & 255) * 32;        // codes
  const int tileN = (t >> 8) * 32;         // out dim e

  const f32x8 zero = {0.f,0.f,0.f,0.f,0.f,0.f,0.f,0.f};
  f32x8 acc[2][2] = {{zero, zero}, {zero, zero}};

  for (int k0 = 0; k0 < DIMD; k0 += 32) {
    FragU ah[2], al[2], bh[2], bl[2];
#pragma unroll
    for (int mi = 0; mi < 2; ++mi) {       // A frag: halves 0-7 -> k=k0+8g+h ; 8-15 -> k=k0+16+8g+(h-8)
      const int row = tileM + mi * 16 + lr;
      const unsigned short* ph = cb_hi + row * DIMD + k0 + 8 * g;
      const unsigned short* pl = cb_lo + row * DIMD + k0 + 8 * g;
      ah[mi].q[0] = *(const uint4v*)ph;  ah[mi].q[1] = *(const uint4v*)(ph + 16);
      al[mi].q[0] = *(const uint4v*)pl;  al[mi].q[1] = *(const uint4v*)(pl + 16);
    }
#pragma unroll
    for (int ni = 0; ni < 2; ++ni) {       // B frag: halves 0-7 -> k=k0+8g+h ; 8-15 -> k=k0+16+8g+(h-8), lane = n
      const int row = tileN + ni * 16 + lr;
      const unsigned short* ph = w_hi + row * DIMD + k0 + 8 * g;
      const unsigned short* pl = w_lo + row * DIMD + k0 + 8 * g;
      bh[ni].q[0] = *(const uint4v*)ph;  bh[ni].q[1] = *(const uint4v*)(ph + 16);
      bl[ni].q[0] = *(const uint4v*)pl;  bl[ni].q[1] = *(const uint4v*)(pl + 16);
    }
#pragma unroll
    for (int mi = 0; mi < 2; ++mi)
#pragma unroll
      for (int ni = 0; ni < 2; ++ni) {
        acc[mi][ni] = wmma_bf16(ah[mi], bh[ni], acc[mi][ni]);
        acc[mi][ni] = wmma_bf16(ah[mi], bl[ni], acc[mi][ni]);
        acc[mi][ni] = wmma_bf16(al[mi], bh[ni], acc[mi][ni]);
      }
  }
#pragma unroll
  for (int mi = 0; mi < 2; ++mi)
#pragma unroll
    for (int ni = 0; ni < 2; ++ni)
#pragma unroll
      for (int v = 0; v < 8; ++v) {        // C layout: lane holds (m = 8g+v, n = lr)
        int row = tileM + mi * 16 + 8 * g + v;
        int col = tileN + ni * 16 + lr;
        float val = acc[mi][ni][v];
        impl[row * DIMD + col]   = val;
        imp_hi[row * DIMD + col] = (unsigned short)(__float_as_uint(val) >> 16);
      }
}

// ---------------------------------------------------------------- c2 norms
__global__ __launch_bounds__(256) void rownorm(const float* __restrict__ m,
                                               float* __restrict__ out) {
  __shared__ float red[256];
  const float* p = m + (size_t)blockIdx.x * DIMD;
  float s = 0.f;
  for (int e = threadIdx.x; e < DIMD; e += 256) { float v = p[e]; s += v * v; }
  red[threadIdx.x] = s; __syncthreads();
  for (int off = 128; off; off >>= 1) {
    if (threadIdx.x < off) red[threadIdx.x] += red[threadIdx.x + off];
    __syncthreads();
  }
  if (threadIdx.x == 0) out[blockIdx.x] = red[0];
}

// ---------------------------------------------------------------- GEMM 2 + fused argmin
// dist-proxy[t,c] = c2[c] - 2 * <x[t], implicit[c]>  (x^2 dropped: constant per token)
__global__ __launch_bounds__(128) void gemm2_argmin(
    const unsigned short* __restrict__ x_hi,
    const unsigned short* __restrict__ imp_hi,
    const float* __restrict__ c2,
    float* __restrict__ partD, int* __restrict__ partI) {
  __shared__ __align__(16) unsigned short xt[32 * XROWS];  // 33,280 B
  const int tokenBase = blockIdx.x * 32;
  const int wave = threadIdx.x >> 5;
  const int lane = threadIdx.x & 31;
  const int g  = lane >> 4;
  const int lr = lane & 15;
  const int chunkSlot = blockIdx.y * 4 + wave;   // 0..31, each wave scans 256 codes

  // stage 32 token rows of bf16 x into LDS (shared by the 4 waves)
  for (int i = threadIdx.x; i < 32 * 64; i += 128) {
    int r = i >> 6, c = i & 63;
    *(uint4v*)(xt + r * XROWS + c * 8) =
        *(const uint4v*)(x_hi + (size_t)(tokenBase + r) * DIMD + c * 8);
  }
  __syncthreads();

  float bestD[16]; int bestI[16];
#pragma unroll
  for (int s = 0; s < 16; ++s) { bestD[s] = 3.4e38f; bestI[s] = 0; }

  const f32x8 zero = {0.f,0.f,0.f,0.f,0.f,0.f,0.f,0.f};
  for (int it = 0; it < 8; ++it) {
    const int cBase = chunkSlot * 256 + it * 32;
    f32x8 acc[2][2] = {{zero, zero}, {zero, zero}};
    for (int k0 = 0; k0 < DIMD; k0 += 32) {
      FragU a[2], b[2];
#pragma unroll
      for (int mi = 0; mi < 2; ++mi) {
        const unsigned short* pa = xt + (mi * 16 + lr) * XROWS + k0 + 8 * g;
        a[mi].q[0] = *(const uint4v*)pa;  a[mi].q[1] = *(const uint4v*)(pa + 16);
      }
#pragma unroll
      for (int ni = 0; ni < 2; ++ni) {
        const unsigned short* pb =
            imp_hi + (size_t)(cBase + ni * 16 + lr) * DIMD + k0 + 8 * g;
        b[ni].q[0] = *(const uint4v*)pb;  b[ni].q[1] = *(const uint4v*)(pb + 16);
      }
#pragma unroll
      for (int mi = 0; mi < 2; ++mi)
#pragma unroll
        for (int ni = 0; ni < 2; ++ni)
          acc[mi][ni] = wmma_bf16(a[mi], b[ni], acc[mi][ni]);
    }
    // fused running-min epilogue (distances never leave registers)
#pragma unroll
    for (int ni = 0; ni < 2; ++ni) {
      const int n = cBase + ni * 16 + lr;
      const float c2n = c2[n];
#pragma unroll
      for (int mi = 0; mi < 2; ++mi)
#pragma unroll
        for (int v = 0; v < 8; ++v) {
          float d = __builtin_fmaf(-2.f, acc[mi][ni][v], c2n);
          int s = mi * 8 + v;
          if (d < bestD[s]) { bestD[s] = d; bestI[s] = n; }
        }
    }
  }
  // reduce across the 16 lanes (n-columns) of each half-wave; first-min tie-break
#pragma unroll
  for (int s = 0; s < 16; ++s) {
    float d = bestD[s]; int idx = bestI[s];
    for (int m = 8; m; m >>= 1) {
      float od = __shfl_xor(d, m, 32);
      int   oi = __shfl_xor(idx, m, 32);
      if (od < d || (od == d && oi < idx)) { d = od; idx = oi; }
    }
    if (lr == 0) {
      int tok = tokenBase + (s >> 3) * 16 + 8 * g + (s & 7);
      partD[tok * 32 + chunkSlot] = d;
      partI[tok * 32 + chunkSlot] = idx;
    }
  }
}

// ---------------------------------------------------------------- finalize per token
__global__ __launch_bounds__(256) void finalize_token(
    const float* __restrict__ x, const float* __restrict__ impl,
    const float* __restrict__ partD, const int* __restrict__ partI,
    float* __restrict__ outQ, float* __restrict__ outIdx,
    float* __restrict__ tokLoss) {
  __shared__ float red[256];
  __shared__ int sIdx;
  const int t = blockIdx.x;
  if (threadIdx.x == 0) {
    float bd = 3.4e38f; int bi = 0x7fffffff;
    for (int s = 0; s < 32; ++s) {     // chunks cover increasing code ranges
      float d = partD[t * 32 + s]; int i = partI[t * 32 + s];
      if (d < bd || (d == bd && i < bi)) { bd = d; bi = i; }
    }
    sIdx = bi;
    outIdx[t] = (float)bi;
  }
  __syncthreads();
  const int idx = sIdx;
  const float* xr = x + (size_t)t * DIMD;
  const float* qr = impl + (size_t)idx * DIMD;
  float s = 0.f;
  for (int e = threadIdx.x; e < DIMD; e += 256) {
    float q = qr[e];
    outQ[(size_t)t * DIMD + e] = q;          // straight-through value == quantized
    float d = xr[e] - q;
    s += d * d;
  }
  red[threadIdx.x] = s; __syncthreads();
  for (int off = 128; off; off >>= 1) {
    if (threadIdx.x < off) red[threadIdx.x] += red[threadIdx.x + off];
    __syncthreads();
  }
  if (threadIdx.x == 0) tokLoss[t] = red[0];
}

__global__ __launch_bounds__(256) void reduce_loss(const float* __restrict__ tokLoss,
                                                   float* __restrict__ outLoss) {
  __shared__ float red[256];
  float s = 0.f;
  for (int i = threadIdx.x; i < NTOK; i += 256) s += tokLoss[i];
  red[threadIdx.x] = s; __syncthreads();
  for (int off = 128; off; off >>= 1) {
    if (threadIdx.x < off) red[threadIdx.x] += red[threadIdx.x + off];
    __syncthreads();
  }
  if (threadIdx.x == 0) outLoss[0] = red[0] * (1.0f / NTOK);
}

// ---------------------------------------------------------------- launch
extern "C" void kernel_launch(void* const* d_in, const int* in_sizes, int n_in,
                              void* d_out, int out_size, void* d_ws, size_t ws_size,
                              hipStream_t stream) {
  const float* x  = (const float*)d_in[0];   // [4,2048,512]
  const float* W  = (const float*)d_in[1];   // [512,512]
  const float* cb = (const float*)d_in[2];   // [8192,512]

  float* out     = (float*)d_out;
  float* outQ    = out;                               // 4*2048*512 f32
  float* outIdx  = out + (size_t)NTOK * DIMD;         // 8192 (indices as f32)
  float* outLoss = outIdx + NTOK;                     // 1

  char* ws = (char*)d_ws;
  size_t off = 0;
  auto alloc = [&](size_t bytes) -> char* {
    char* p = ws + off; off += (bytes + 255) & ~(size_t)255; return p;
  };
  unsigned short* x_hi   = (unsigned short*)alloc((size_t)NTOK  * DIMD * 2);
  unsigned short* cb_hi  = (unsigned short*)alloc((size_t)NCODE * DIMD * 2);
  unsigned short* cb_lo  = (unsigned short*)alloc((size_t)NCODE * DIMD * 2);
  unsigned short* w_hi   = (unsigned short*)alloc((size_t)DIMD  * DIMD * 2);
  unsigned short* w_lo   = (unsigned short*)alloc((size_t)DIMD  * DIMD * 2);
  float*          impl   = (float*)         alloc((size_t)NCODE * DIMD * 4);
  unsigned short* imp_hi = (unsigned short*)alloc((size_t)NCODE * DIMD * 2);
  float*          c2     = (float*)         alloc((size_t)NCODE * 4);
  float*          partD  = (float*)         alloc((size_t)NTOK * 32 * 4);
  int*            partI  = (int*)           alloc((size_t)NTOK * 32 * 4);
  float*          tokLoss= (float*)         alloc((size_t)NTOK * 4);

  split_f32_to_bf16<<<512, 256, 0, stream>>>(x,  x_hi,  nullptr, NTOK * DIMD);
  split_f32_to_bf16<<<512, 256, 0, stream>>>(cb, cb_hi, cb_lo,   NCODE * DIMD);
  split_f32_to_bf16<<<256, 256, 0, stream>>>(W,  w_hi,  w_lo,    DIMD * DIMD);

  gemm1_implicit<<<1024, 128, 0, stream>>>(cb_hi, cb_lo, w_hi, w_lo, impl, imp_hi);
  rownorm<<<NCODE, 256, 0, stream>>>(impl, c2);

  dim3 g2(256, 8);  // 256 token tiles x 32 code chunks (4 waves/block)
  gemm2_argmin<<<g2, 128, 0, stream>>>(x_hi, imp_hi, c2, partD, partI);

  finalize_token<<<NTOK, 256, 0, stream>>>(x, impl, partD, partI, outQ, outIdx, tokLoss);
  reduce_loss<<<1, 256, 0, stream>>>(tokLoss, outLoss);
}
